// MultiHeadAttention_19404662243358
// MI455X (gfx1250) — matrix operand order repaired and checked by reading, not compiled, not run on hardware
//
#include <hip/hip_runtime.h>
#include <hip/hip_bf16.h>
#include <stdint.h>

#define B_  8
#define L_  1024
#define D_  1024
#define H_  16
#define DK_ 64
#define BL_ (B_ * L_)
#define SLD (L_ + 4)   // padded S row stride (floats); +16B keeps 16B alignment

typedef __bf16 bf16_t;
typedef __attribute__((ext_vector_type(16))) __bf16 v16bf;
typedef __attribute__((ext_vector_type(8)))  __bf16 v8bf;
typedef __attribute__((ext_vector_type(2)))  __bf16 v2bf;
typedef __attribute__((ext_vector_type(8)))  float  v8f;
typedef __attribute__((ext_vector_type(4)))  float  v4f;
typedef __attribute__((ext_vector_type(2)))  float  v2f;
typedef __attribute__((ext_vector_type(4)))  uint32_t v4u;
typedef __attribute__((ext_vector_type(4)))  int    v4i;

#if defined(__AMDGCN__) && __has_builtin(__builtin_amdgcn_global_load_async_to_lds_b128) && __has_builtin(__builtin_amdgcn_s_wait_asynccnt)
#define ASYNC_LDS 1
#else
#define ASYNC_LDS 0
#endif

static __device__ inline v16bf cat8(v8bf lo, v8bf hi) {
  return __builtin_shufflevector(lo, hi, 0,1,2,3,4,5,6,7,8,9,10,11,12,13,14,15);
}

// A fragment (16x32 bf16, MxK) from row-major LDS tile; ISA layout:
// lane 0-15: row=lane, K = {0..7, 16..23}; lane 16-31: row=lane-16, K = {8..15, 24..31}
static __device__ inline v16bf frag_a(const bf16_t* base, int ldr, int lane) {
  const int row = lane & 15, kh = lane >> 4;
  const bf16_t* p = base + row * ldr;
  v8bf lo = *(const v8bf*)(p + kh * 8);
  v8bf hi = *(const v8bf*)(p + 16 + kh * 8);
  return cat8(lo, hi);
}

// B fragment (32x16 bf16, KxN) from LDS tile stored transposed (N rows x K cols):
// lane 0-15: col=lane, K = {0..7, 16..23}; lane 16-31: col=lane-16, K = {8..15, 24..31}
static __device__ inline v16bf frag_b(const bf16_t* base, int ldr, int lane) {
  const int n = lane & 15, kh = lane >> 4;
  const bf16_t* p = base + n * ldr;
  v8bf lo = *(const v8bf*)(p + kh * 8);
  v8bf hi = *(const v8bf*)(p + 16 + kh * 8);
  return cat8(lo, hi);
}

// A fragment gathered from an fp32 LDS tile, converting to bf16 in registers.
static __device__ inline v16bf frag_a_f32(const float* base, int ldr, int lane) {
  const int row = lane & 15, kh = lane >> 4;
  const float* p = base + row * ldr;
  v4f a0 = *(const v4f*)(p + kh * 8);
  v4f a1 = *(const v4f*)(p + kh * 8 + 4);
  v4f a2 = *(const v4f*)(p + 16 + kh * 8);
  v4f a3 = *(const v4f*)(p + 16 + kh * 8 + 4);
  v16bf r;
#pragma unroll
  for (int i = 0; i < 4; ++i) {
    r[i]      = (bf16_t)a0[i];
    r[4 + i]  = (bf16_t)a1[i];
    r[8 + i]  = (bf16_t)a2[i];
    r[12 + i] = (bf16_t)a3[i];
  }
  return r;
}

static __device__ inline v8f wmma_bf16(v16bf a, v16bf b, v8f c) {
  return __builtin_amdgcn_wmma_f32_16x16x32_bf16(false, a, false, b, (short)0, c,
                                                 false, false);
}

static __device__ inline v2bf pack_bf16(float x, float y) {
  v2bf t;
  t[0] = (bf16_t)x;
  t[1] = (bf16_t)y;
  return t;
}

#if ASYNC_LDS
// Per-lane async copy of 16B global -> LDS (tracked by ASYNCcnt).
typedef __attribute__((address_space(1))) v4i gas_v4i;
typedef __attribute__((address_space(3))) v4i las_v4i;
static __device__ inline void async_cp16(const void* g, void* l) {
  __builtin_amdgcn_global_load_async_to_lds_b128((gas_v4i*)g, (las_v4i*)l, 0, 0);
}
#endif

// ---------------------------------------------------------------------------
// Per-head projection: Y[h] = X @ W[h],  X:[BL,D] fp32, W:[H,D,DK] fp32,
// Y:[H,BL,DK] bf16.  Block tile 128x64, Kc=32, 8 waves as 4x2, 32x32 per wave.
// ---------------------------------------------------------------------------
__global__ __launch_bounds__(256)
void proj_head_gemm(const float* __restrict__ X, const float* __restrict__ W,
                    bf16_t* __restrict__ Y) {
  __shared__ union {
    struct { bf16_t A[128][32]; bf16_t Bm[64][32]; } st;   // 12 KB staging
    bf16_t outT[128 * 64];                                  // 16 KB output tile
  } u;
  const int tid = threadIdx.x, lane = tid & 31, w = tid >> 5;
  const int wrow = w >> 1, wcol = w & 1;
  const int m0 = blockIdx.x * 128;
  const int h  = blockIdx.y;
  const float* Wh = W + (size_t)h * D_ * DK_;

  v8f c[2][2] = {};
  for (int k0 = 0; k0 < D_; k0 += 32) {
#pragma unroll
    for (int i = 0; i < 8; ++i) {       // stage A: float2 load -> packed bf16x2
      int e = tid + i * 256;            // 128 rows x 16 pairs
      int r = e >> 4, kp = e & 15;
      v2f x = *(const v2f*)(X + (size_t)(m0 + r) * D_ + k0 + kp * 2);
      *(v2bf*)&u.st.A[r][kp * 2] = pack_bf16(x[0], x[1]);
    }
#pragma unroll
    for (int i = 0; i < 4; ++i) {       // stage B transposed: sB[n][k] = W[k][n]
      int e = tid + i * 256;            // 64 n x 16 k-pairs
      int n = e & 63, kp = e >> 6;
      float x0 = Wh[(size_t)(k0 + kp * 2) * DK_ + n];
      float x1 = Wh[(size_t)(k0 + kp * 2 + 1) * DK_ + n];
      *(v2bf*)&u.st.Bm[n][kp * 2] = pack_bf16(x0, x1);
    }
    __syncthreads();
    v16bf a0 = frag_a(&u.st.A[wrow * 32][0], 32, lane);
    v16bf a1 = frag_a(&u.st.A[wrow * 32 + 16][0], 32, lane);
    v16bf b0 = frag_b(&u.st.Bm[wcol * 32][0], 32, lane);
    v16bf b1 = frag_b(&u.st.Bm[wcol * 32 + 16][0], 32, lane);
    c[0][0] = wmma_bf16(a0, b0, c[0][0]);
    c[0][1] = wmma_bf16(a0, b1, c[0][1]);
    c[1][0] = wmma_bf16(a1, b0, c[1][0]);
    c[1][1] = wmma_bf16(a1, b1, c[1][1]);
    __syncthreads();
  }
  // Bounce D fragments through LDS, then stream the contiguous 16KB tile out.
  const int kh = lane >> 4, n16 = lane & 15;
#pragma unroll
  for (int i = 0; i < 2; ++i)
#pragma unroll
    for (int j = 0; j < 2; ++j)
#pragma unroll
      for (int r = 0; r < 8; ++r) {
        int nn = wcol * 32 + j * 16 + n16;
        int mm = wrow * 32 + i * 16 + kh * 8 + r;
        u.outT[mm * 64 + nn] = (bf16_t)c[i][j][r];
      }
  __syncthreads();
  bf16_t* Yb = Y + ((size_t)h * BL_ + m0) * DK_;   // tile is contiguous in Y
#pragma unroll
  for (int i = 0; i < 4; ++i) {
    int e = tid + i * 256;                          // 1024 x 16B chunks
    *(v4u*)((char*)Yb + e * 16) = *(const v4u*)((const char*)u.outT + e * 16);
  }
}

// ---------------------------------------------------------------------------
// Fused scores + mask + softmax + PV. One block = (head*batch, 32 q-rows).
// S[32][SLD] fp32 in dynamic LDS; K tiles double-buffered via async loads;
// attn written once to d_out; P never leaves LDS before the PV WMMA.
// ---------------------------------------------------------------------------
__global__ __launch_bounds__(256)
void attn_softmax_pv(const bf16_t* __restrict__ Qh, const bf16_t* __restrict__ Kh,
                     const bf16_t* __restrict__ Vh,
                     const unsigned char* __restrict__ mask,
                     float* __restrict__ attn_out,   // [H*B, L, L]
                     bf16_t* __restrict__ O) {       // [BL, D] head-major cols
  extern __shared__ char smem[];
  float*  S   = (float*)smem;                                   // [32][SLD]
  bf16_t* sQ  = (bf16_t*)(smem + 32 * SLD * 4);                 // [32][64] (4KB)
  bf16_t* sT0 = (bf16_t*)(smem + 32 * SLD * 4 + 32 * DK_ * 2);  // [64][64] (8KB)
  bf16_t* sT1 = sT0 + 64 * 64;                                  // [64][64] (8KB)

  const int tid = threadIdx.x, lane = tid & 31, w = tid >> 5;
  const int q0 = blockIdx.x * 32;
  const int hb = blockIdx.y;
  const int h = hb / B_, b = hb % B_;
  const size_t head_row0 = (size_t)h * BL_ + (size_t)b * L_;
  const char* qsrc = (const char*)(Qh + (head_row0 + q0) * DK_);
  const char* ksrc0 = (const char*)(Kh + head_row0 * DK_);

  // ---- prologue: Q block (4KB) + K tile 0, then hoist Q A-fragments ----
#if ASYNC_LDS
  async_cp16(qsrc + tid * 16, (char*)sQ + tid * 16);
#pragma unroll
  for (int i = 0; i < 2; ++i)
    async_cp16(ksrc0 + (tid + i * 256) * 16, (char*)sT0 + (tid + i * 256) * 16);
  __builtin_amdgcn_s_wait_asynccnt(2);   // Q complete (in-order), K0 in flight
#else
  *(v4u*)((char*)sQ + tid * 16) = *(const v4u*)(qsrc + tid * 16);
#pragma unroll
  for (int i = 0; i < 2; ++i)
    *(v4u*)((char*)sT0 + (tid + i * 256) * 16) =
        *(const v4u*)(ksrc0 + (tid + i * 256) * 16);
#endif
  __syncthreads();

  const int wr1 = w >> 2, wc1 = w & 3;   // 2 x 4 waves over 32x64
  v16bf aq[2];
  aq[0] = frag_a(sQ + (wr1 * 16) * DK_, DK_, lane);
  aq[1] = frag_a(sQ + (wr1 * 16) * DK_ + 32, DK_, lane);

  const float scale = 0.03125f;  // 1/sqrt(D) — reference scales by sqrt(model_dim)

  // ---- phase 1: S = Q K^T * scale, K tiles double-buffered ----
  for (int it = 0; it < L_ / 64; ++it) {
    bf16_t* cur = (it & 1) ? sT1 : sT0;
#if ASYNC_LDS
    if (it < L_ / 64 - 1) {   // issue next tile into the other buffer
      bf16_t* nxt = (it & 1) ? sT0 : sT1;
      const char* ksrc = ksrc0 + (size_t)(it + 1) * 64 * DK_ * 2;
#pragma unroll
      for (int i = 0; i < 2; ++i)
        async_cp16(ksrc + (tid + i * 256) * 16, (char*)nxt + (tid + i * 256) * 16);
      __builtin_amdgcn_s_wait_asynccnt(2);   // current tile complete
    } else {
      __builtin_amdgcn_s_wait_asynccnt(0);
    }
    __syncthreads();
#else
    if (it > 0) {
      const char* ksrc = ksrc0 + (size_t)it * 64 * DK_ * 2;
#pragma unroll
      for (int i = 0; i < 2; ++i)
        *(v4u*)((char*)cur + (tid + i * 256) * 16) =
            *(const v4u*)(ksrc + (tid + i * 256) * 16);
    }
    __syncthreads();
#endif
    v8f c = {};
#pragma unroll
    for (int ks = 0; ks < 2; ++ks) {
      v16bf bm = frag_b(cur + (wc1 * 16) * DK_ + ks * 32, DK_, lane);
      c = wmma_bf16(aq[ks], bm, c);
    }
    const int n16 = lane & 15, kh = lane >> 4;
#pragma unroll
    for (int r = 0; r < 8; ++r) {
      int qq = wr1 * 16 + kh * 8 + r;
      int ss = it * 64 + wc1 * 16 + n16;
      S[qq * SLD + ss] = c[r] * scale;
    }
    __syncthreads();   // all waves done reading `cur` before it is re-filled
  }

  // ---- phase 2: mask + row softmax (4 rows/wave, 4-wide vectors, row kept
  //      in registers between passes) ----
  for (int r = 0; r < 4; ++r) {
    int row = w * 4 + r;
    float* Srow = S + row * SLD;
    const uint32_t* mrow =
        (const uint32_t*)(mask + ((size_t)b * L_ + (q0 + row)) * L_);
    v4f xv[8];
    float mx = -3.0e38f;
#pragma unroll
    for (int i = 0; i < 8; ++i) {
      int s4 = lane + i * 32;                 // group of 4 columns
      v4f x = *(const v4f*)(Srow + s4 * 4);
      uint32_t mb = mrow[s4];
#pragma unroll
      for (int j = 0; j < 4; ++j) {
        if ((mb >> (8 * j)) & 0xffu) x[j] = -1.0e18f;
        mx = fmaxf(mx, x[j]);
      }
      xv[i] = x;
    }
#pragma unroll
    for (int off = 16; off > 0; off >>= 1) mx = fmaxf(mx, __shfl_xor(mx, off, 32));
    float sum = 0.f;
#pragma unroll
    for (int i = 0; i < 8; ++i) {
      v4f e;
#pragma unroll
      for (int j = 0; j < 4; ++j) {
        e[j] = __expf(xv[i][j] - mx);
        sum += e[j];
      }
      xv[i] = e;
    }
#pragma unroll
    for (int off = 16; off > 0; off >>= 1) sum += __shfl_xor(sum, off, 32);
    float inv = 1.0f / sum;
    float* arow = attn_out + ((size_t)hb * L_ + (q0 + row)) * L_;
#pragma unroll
    for (int i = 0; i < 8; ++i) {
      int s4 = lane + i * 32;
      v4f p = xv[i] * inv;
      *(v4f*)(Srow + s4 * 4) = p;   // normalized P stays in LDS for PV
      *(v4f*)(arow + s4 * 4) = p;   // mandatory attn output, written once, b128
    }
  }
  __syncthreads();

  // ---- phase 3: O = P V.  V^T tiles register-prefetched ahead of WMMA. ----
  const int wr3 = w >> 2, wc3 = w & 3;
  v8f co = {};
  v2bf tv[4];
#pragma unroll
  for (int i = 0; i < 4; ++i) {            // prefetch V tile 0
    int e = tid + i * 256, sp = e >> 6, vv = e & 63;
    tv[i][0] = Vh[(head_row0 + sp * 2) * DK_ + vv];
    tv[i][1] = Vh[(head_row0 + sp * 2 + 1) * DK_ + vv];
  }
  for (int s0 = 0; s0 < L_; s0 += 32) {
#pragma unroll
    for (int i = 0; i < 4; ++i) {          // commit prefetched tile to LDS
      int e = tid + i * 256, sp = e >> 6, vv = e & 63;
      *(v2bf*)&sT0[vv * 32 + sp * 2] = tv[i];
    }
    __syncthreads();
    if (s0 + 32 < L_) {
#pragma unroll
      for (int i = 0; i < 4; ++i) {        // prefetch next tile during WMMA
        int e = tid + i * 256, sp = e >> 6, vv = e & 63;
        tv[i][0] = Vh[(head_row0 + s0 + 32 + sp * 2) * DK_ + vv];
        tv[i][1] = Vh[(head_row0 + s0 + 32 + sp * 2 + 1) * DK_ + vv];
      }
    }
    v16bf a  = frag_a_f32(S + (wr3 * 16) * SLD + s0, SLD, lane);
    v16bf bv = frag_b(sT0 + (wc3 * 16) * 32, 32, lane);
    co = wmma_bf16(a, bv, co);
    __syncthreads();
  }
  { // bounce O tile (32x64 bf16 = 4KB) through sQ, then b128 stores
    const int n16 = lane & 15, kh = lane >> 4;
#pragma unroll
    for (int r = 0; r < 8; ++r) {
      int qq = wr3 * 16 + kh * 8 + r;
      int vv = wc3 * 16 + n16;
      sQ[qq * 64 + vv] = (bf16_t)co[r];
    }
    __syncthreads();
    int row = tid >> 3, chunk = tid & 7;   // 32 rows x 8 chunks of 8 bf16
    bf16_t* dst = O + ((size_t)(b * L_ + q0 + row)) * D_ + h * DK_ + chunk * 8;
    *(v4u*)dst = *(const v4u*)(sQ + row * 64 + chunk * 8);
  }
}

// ---------------------------------------------------------------------------
// out = O @ proj_w^T + proj_b.  proj_w rows are exactly the NxK LDS layout.
// ---------------------------------------------------------------------------
__global__ __launch_bounds__(256)
void out_proj_gemm(const bf16_t* __restrict__ O, const float* __restrict__ Wp,
                   const float* __restrict__ bias, float* __restrict__ out) {
  __shared__ bf16_t sA[128][32];
  __shared__ bf16_t sB[64][32];
  const int tid = threadIdx.x, lane = tid & 31, w = tid >> 5;
  const int wrow = w >> 1, wcol = w & 1;
  const int m0 = blockIdx.x * 128;
  const int n0 = blockIdx.y * 64;

  v8f c[2][2] = {};
  for (int k0 = 0; k0 < D_; k0 += 32) {
#pragma unroll
    for (int i = 0; i < 2; ++i) {       // stage A: b128 bf16 copies
      int e = tid + i * 256;            // 128 rows x 4 chunks of 8 bf16
      int r = e >> 2, c4 = e & 3;
      *(v4u*)&sA[r][c4 * 8] =
          *(const v4u*)(O + (size_t)(m0 + r) * D_ + k0 + c4 * 8);
    }
#pragma unroll
    for (int i = 0; i < 4; ++i) {       // stage B: sB[n][k] = proj_w[n0+n][k]
      int e = tid + i * 256;            // 64 n x 16 k-pairs
      int n = e >> 4, kp = e & 15;
      v2f x = *(const v2f*)(Wp + (size_t)(n0 + n) * D_ + k0 + kp * 2);
      *(v2bf*)&sB[n][kp * 2] = pack_bf16(x[0], x[1]);
    }
    __syncthreads();
    v16bf a0 = frag_a(&sA[wrow * 32][0], 32, lane);
    v16bf a1 = frag_a(&sA[wrow * 32 + 16][0], 32, lane);
    v16bf b0 = frag_b(&sB[wcol * 32][0], 32, lane);
    v16bf b1 = frag_b(&sB[wcol * 32 + 16][0], 32, lane);
    c[0][0] = wmma_bf16(a0, b0, c[0][0]);
    c[0][1] = wmma_bf16(a0, b1, c[0][1]);
    c[1][0] = wmma_bf16(a1, b0, c[1][0]);
    c[1][1] = wmma_bf16(a1, b1, c[1][1]);
    __syncthreads();
  }
  const int kh = lane >> 4, n16 = lane & 15;
#pragma unroll
  for (int i = 0; i < 2; ++i)
#pragma unroll
    for (int j = 0; j < 2; ++j)
#pragma unroll
      for (int r = 0; r < 8; ++r) {
        int nn = n0 + wcol * 32 + j * 16 + n16;
        int mm = m0 + wrow * 32 + i * 16 + kh * 8 + r;
        out[(size_t)mm * D_ + nn] = c[i][j][r] + bias[nn];
      }
}

extern "C" void kernel_launch(void* const* d_in, const int* in_sizes, int n_in,
                              void* d_out, int out_size, void* d_ws, size_t ws_size,
                              hipStream_t stream) {
  const float* q      = (const float*)d_in[0];
  const float* k      = (const float*)d_in[1];
  const float* v      = (const float*)d_in[2];
  const unsigned char* mask = (const unsigned char*)d_in[3];
  const float* w_qs   = (const float*)d_in[4];
  const float* w_ks   = (const float*)d_in[5];
  const float* w_vs   = (const float*)d_in[6];
  const float* proj_w = (const float*)d_in[7];
  const float* proj_b = (const float*)d_in[8];

  float* out   = (float*)d_out;
  float* attns = out + (size_t)BL_ * D_;   // [H*B, L, L] after out[B,L,D]

  const size_t headElems = (size_t)H_ * BL_ * DK_;
  bf16_t* Qh = (bf16_t*)d_ws;
  bf16_t* Kh = Qh + headElems;
  bf16_t* Vh = Kh + headElems;
  bf16_t* O  = Vh + headElems;             // [BL, D] bf16, head-major features

  dim3 blk(256);
  dim3 gproj(BL_ / 128, H_);
  proj_head_gemm<<<gproj, blk, 0, stream>>>(q, w_qs, Qh);
  proj_head_gemm<<<gproj, blk, 0, stream>>>(k, w_ks, Kh);
  proj_head_gemm<<<gproj, blk, 0, stream>>>(v, w_vs, Vh);

  dim3 gattn(L_ / 32, H_ * B_);
  size_t smem = (size_t)32 * SLD * 4 + 32 * DK_ * 2 + 2 * 64 * 64 * 2;  // ~148 KB
  attn_softmax_pv<<<gattn, blk, smem, stream>>>(Qh, Kh, Vh, mask, attns, O);

  dim3 gout(BL_ / 128, D_ / 64);
  out_proj_gemm<<<gout, blk, 0, stream>>>(O, proj_w, proj_b, out);
}
